// Attention_15315853377620
// MI455X (gfx1250) — matrix operand order repaired and checked by reading, not compiled, not run on hardware
//
#include <hip/hip_runtime.h>

// Problem sizes
#define B_SZ   256
#define H_SZ   128
#define N_SZ   1000
#define K_SZ   512            // 4*H
#define NT     63             // ceil(N/16)
#define NCHUNK 4
#define TPC    16             // n-tiles per chunk (last chunk does 15)

// LDS layout (dynamic shared memory)
#define WOFF      0
#define WBYTES    (8*16*32*32)          // 131072: [htile][kstep][lane][16 bf16]
#define HIDSTRIDE 520                   // bf16 elems per n-row (512 + pad for banks)
#define HIDOFF    WBYTES
#define HIDBYTES  (16*HIDSTRIDE*2)      // 16640
#define ATTNOFF   (HIDOFF + HIDBYTES)
#define SMEM_BYTES (ATTNOFF + 64)

typedef __bf16        v16bf __attribute__((ext_vector_type(16)));
typedef float         v8f   __attribute__((ext_vector_type(8)));
typedef float         f32x4 __attribute__((ext_vector_type(4)));
typedef unsigned int  u32x4 __attribute__((ext_vector_type(4)));

union Frag { u32x4 q[2]; v16bf v; };    // 32 bytes, POD for clean b128 loads

__device__ __forceinline__ unsigned short f2bf(float f) {
    union { float f; unsigned u; } x; x.f = f;
    unsigned r = x.u + 0x7FFFu + ((x.u >> 16) & 1u);   // round-to-nearest-even
    return (unsigned short)(r >> 16);
}

// ---------------------------------------------------------------------------
// Kernel 1: logits[b][n] = sum_h v[h] * tanh(sum_k W[h][k] * hidden[b][k][n])
// grid = (NCHUNK, B), block = 256 (8 wave32)
// ---------------------------------------------------------------------------
__global__ __launch_bounds__(256)
void attn_logits_kernel(const float* __restrict__ adj,
                        const float* __restrict__ sta,
                        const float* __restrict__ dyn,
                        const float* __restrict__ dec,
                        const float* __restrict__ v,
                        const float* __restrict__ W,
                        float* __restrict__ logits)
{
    extern __shared__ unsigned char smem[];
    unsigned*       Wlds = (unsigned*)(smem + WOFF);
    unsigned short* hidT = (unsigned short*)(smem + HIDOFF);    // [16 n][520 k]
    float*          attn = (float*)(smem + ATTNOFF);            // [16]

    const int tid   = threadIdx.x;
    const int b     = blockIdx.y;
    const int chunk = blockIdx.x;
    const int wv    = tid >> 5;          // wave id 0..7  -> h-tile
    const int lane  = tid & 31;
    const int hif   = (lane >> 4) & 1;   // lane-half select
    const int nloc  = lane & 15;

    // ---- Stage W -> LDS bf16, pre-swizzled to A-fragment layout ----
    // A 16x32 bf16 layout: lane m=l&15; k = 32s + (e>=8?16:0) + (l&16?8:0) + (e&7)
#pragma unroll 4
    for (int it = 0; it < 128; ++it) {
        int idx = it * 256 + tid;                 // ((w*16+s)*32+l)*8 + j
        int j = idx & 7;
        int l = (idx >> 3) & 31;
        int s = (idx >> 8) & 15;
        int w = idx >> 12;
        int h = (w << 4) + (l & 15);
        int e = 2 * j;
        int k = (s << 5) + ((e & 8) ? 16 : 0) + ((l & 16) ? 8 : 0) + (e & 7);
        float f0 = W[h * K_SZ + k];
        float f1 = W[h * K_SZ + k + 1];
        Wlds[idx] = (unsigned)f2bf(f0) | ((unsigned)f2bf(f1) << 16);
    }

    // ---- Stage decoder rows (k = 384..511), n-invariant for this b ----
#pragma unroll
    for (int it = 0; it < 4; ++it) {
        int idx = it * 256 + tid;       // 0..1023 = 64 k-pairs * 16 n
        int n  = idx & 15;
        int kp = idx >> 4;              // 0..63
        int k  = 384 + 2 * kp;
        float f0 = dec[b * H_SZ + 2 * kp];
        float f1 = dec[b * H_SZ + 2 * kp + 1];
        unsigned pk = (unsigned)f2bf(f0) | ((unsigned)f2bf(f1) << 16);
        *(unsigned*)&hidT[n * HIDSTRIDE + k] = pk;
    }

    // ---- Preload v with C-matrix row mapping: h = 16*wv + r + 8*hif ----
    float vreg[8];
#pragma unroll
    for (int r = 0; r < 8; ++r)
        vreg[r] = v[(wv << 4) + r + (hif << 3)];

    const int t0 = chunk * TPC;
    const int t1 = (t0 + TPC < NT) ? (t0 + TPC) : NT;

    for (int t = t0; t < t1; ++t) {
        const int n0 = t << 4;

        // ---- Stage hidden tile: rows k=0..383 from 3 tensors, transpose to
        //      hidT[n][k] bf16 (pairs of k-rows packed into one u32 store) ----
#pragma unroll
        for (int it = 0; it < 3; ++it) {
            int item = it * 256 + tid;            // 0..767 = 192 k-pairs * 4 quads
            int kp = item >> 2;                   // k-pair index 0..191
            int q  = item & 3;                    // float4 quad along n
            int k  = 2 * kp;
            const float* src; int kk = k;
            if (k < 128)      { src = adj; }
            else if (k < 256) { src = sta; kk = k - 128; }
            else              { src = dyn; kk = k - 256; }
            int n = n0 + 4 * q;
            f32x4 a = (f32x4)0.0f, c = (f32x4)0.0f;
            if (n + 3 < N_SZ) {     // N%4==0, so float4s are all-or-nothing
                a = *(const f32x4*)(src + (size_t)(b * H_SZ + kk)     * N_SZ + n);
                c = *(const f32x4*)(src + (size_t)(b * H_SZ + kk + 1) * N_SZ + n);
            }
#pragma unroll
            for (int i = 0; i < 4; ++i) {
                unsigned pk = (unsigned)f2bf(a[i]) | ((unsigned)f2bf(c[i]) << 16);
                *(unsigned*)&hidT[(4 * q + i) * HIDSTRIDE + k] = pk;
            }
        }
        if (tid < 16) attn[tid] = 0.0f;
        __syncthreads();

        // ---- WMMA: scores tile [16 h x 16 n], K=512 in 16 steps of 32 ----
        v8f acc = {};
#pragma unroll
        for (int s = 0; s < 16; ++s) {
            Frag A, Bf;
            const unsigned char* ap = smem + (((wv * 16 + s) * 32 + lane) * 32);
            A.q[0] = *(const u32x4*)(ap);
            A.q[1] = *(const u32x4*)(ap + 16);
            const unsigned char* bp = smem + HIDOFF +
                (size_t)(nloc * HIDSTRIDE + s * 32 + hif * 8) * 2;
            Bf.q[0] = *(const u32x4*)(bp);
            Bf.q[1] = *(const u32x4*)(bp + 32);
            acc = __builtin_amdgcn_wmma_f32_16x16x32_bf16(
                      false, A.v, false, Bf.v, (short)0, acc, false, false);
        }

        // ---- epilogue: logit contribution = sum_r tanh(acc[r]) * v[h(r)] ----
        float p = 0.0f;
#pragma unroll
        for (int r = 0; r < 8; ++r)
            p += tanhf(acc[r]) * vreg[r];
        atomicAdd(&attn[nloc], p);
        __syncthreads();

        if (tid < 16) {
            int n = n0 + tid;
            if (n < N_SZ) logits[(size_t)b * N_SZ + n] = attn[tid];
        }
        __syncthreads();   // protect attn/hidT before next tile's staging
    }
}

// ---------------------------------------------------------------------------
// Kernel 2: row softmax over N=1000.  grid = B, block = 256
// ---------------------------------------------------------------------------
__global__ __launch_bounds__(256)
void softmax_kernel(const float* __restrict__ logits, float* __restrict__ out)
{
    __shared__ float red[256];
    const int b = blockIdx.x, tid = threadIdx.x;

    float x[4];
    float m = -3.402823466e38f;
#pragma unroll
    for (int i = 0; i < 4; ++i) {
        int n = tid + i * 256;
        x[i] = (n < N_SZ) ? logits[(size_t)b * N_SZ + n] : -3.402823466e38f;
        m = fmaxf(m, x[i]);
    }
    red[tid] = m; __syncthreads();
    for (int off = 128; off > 0; off >>= 1) {
        if (tid < off) red[tid] = fmaxf(red[tid], red[tid + off]);
        __syncthreads();
    }
    m = red[0]; __syncthreads();

    float e[4];
    float s = 0.0f;
#pragma unroll
    for (int i = 0; i < 4; ++i) {
        int n = tid + i * 256;
        e[i] = (n < N_SZ) ? __expf(x[i] - m) : 0.0f;
        s += e[i];
    }
    red[tid] = s; __syncthreads();
    for (int off = 128; off > 0; off >>= 1) {
        if (tid < off) red[tid] += red[tid + off];
        __syncthreads();
    }
    float inv = 1.0f / red[0];
#pragma unroll
    for (int i = 0; i < 4; ++i) {
        int n = tid + i * 256;
        if (n < N_SZ) out[(size_t)b * N_SZ + n] = e[i] * inv;
    }
}

// ---------------------------------------------------------------------------
extern "C" void kernel_launch(void* const* d_in, const int* in_sizes, int n_in,
                              void* d_out, int out_size, void* d_ws, size_t ws_size,
                              hipStream_t stream)
{
    const float* adj = (const float*)d_in[0];   // [B,H,N]
    const float* sta = (const float*)d_in[1];   // [B,H,N]
    const float* dyn = (const float*)d_in[2];   // [B,H,N]
    const float* dec = (const float*)d_in[3];   // [B,H]
    const float* v   = (const float*)d_in[4];   // [1,1,H]
    const float* W   = (const float*)d_in[5];   // [1,H,4H]
    float* logits = (float*)d_ws;               // B*N floats scratch
    float* out    = (float*)d_out;              // [B,1,N]

    dim3 grid(NCHUNK, B_SZ);
    attn_logits_kernel<<<grid, dim3(256), SMEM_BYTES, stream>>>(
        adj, sta, dyn, dec, v, W, logits);
    softmax_kernel<<<dim3(B_SZ), dim3(256), 0, stream>>>(logits, out);
}
